// MultiHeadAttention_62491774157344
// MI455X (gfx1250) — matrix operand order repaired and checked by reading, not compiled, not run on hardware
//
#include <hip/hip_runtime.h>
#include <hip/hip_fp16.h>

#define NH 16
#define DMODEL 1024
#define DK 64
#define BB 2
#define SS 2048

typedef __attribute__((ext_vector_type(16))) _Float16 v16h;
typedef __attribute__((ext_vector_type(8)))  _Float16 v8h;
typedef __attribute__((ext_vector_type(8)))  float    v8f;

__device__ __forceinline__ v8f wmma_f16(v16h a, v16h b, v8f c) {
  return __builtin_amdgcn_wmma_f32_16x16x32_f16(false, a, false, b, (short)0, c, false, false);
}

// CDNA5 async DMA: global -> LDS, 16 bytes per lane, tracked by ASYNCcnt.
__device__ __forceinline__ void async_copy_b128(void* lds_dst, const void* gsrc) {
  const unsigned lds = (unsigned)(uintptr_t)lds_dst;   // LDS aperture: low 32 bits = LDS offset
  asm volatile("global_load_async_to_lds_b128 %0, %1, off"
               :: "v"(lds), "v"(gsrc) : "memory");
}
__device__ __forceinline__ void wait_async() {
  asm volatile("s_wait_asynccnt 0x0" ::: "memory");
}

// A-matrix 16x32 f16 fragment (ISA 7.12.2): two contiguous 8-half runs -> ds_load_b128 x2.
__device__ __forceinline__ v16h load_a_frag(const _Float16* p, int ld) {
  const int lane = threadIdx.x & 31;
  const int m = lane & 15, hi = lane >> 4;
  const _Float16* r0 = p + m * ld + hi * 8;
  v16h a;
#pragma unroll
  for (int e = 0; e < 8; ++e) a[e] = r0[e];
#pragma unroll
  for (int e = 0; e < 8; ++e) a[8 + e] = r0[16 + e];
  return a;
}

// B-matrix 32x16 fragment from an n-major tile: B[k][n] = p[n*ld + k]; two 8-half runs (k=8h.., 16+8h..).
__device__ __forceinline__ v16h load_bn_frag(const _Float16* p, int ld) {
  const int lane = threadIdx.x & 31;
  const int n = lane & 15, hi = lane >> 4;
  const _Float16* r0 = p + n * ld + hi * 8;
  v16h b;
#pragma unroll
  for (int e = 0; e < 8; ++e) b[e] = r0[e];
#pragma unroll
  for (int e = 0; e < 8; ++e) b[8 + e] = r0[16 + e];
  return b;
}

__global__ void cvt_f32_f16(const float* __restrict__ in, _Float16* __restrict__ out, int n) {
  for (int i = blockIdx.x * blockDim.x + threadIdx.x; i < n; i += gridDim.x * blockDim.x)
    out[i] = (_Float16)in[i];
}

// Tiled transpose + f32->f16: in [nmat][rows][cols] -> out [nmat][cols][rows]
__global__ __launch_bounds__(256) void cvt_transpose(
    const float* __restrict__ in, _Float16* __restrict__ out, int rows, int cols) {
  __shared__ float tile[32][33];
  const int m = blockIdx.z;
  const int r0 = blockIdx.y * 32, c0 = blockIdx.x * 32;
  const int tx = threadIdx.x & 31, ty = threadIdx.x >> 5;
  const float* src = in + (size_t)m * rows * cols;
  _Float16* dst = out + (size_t)m * rows * cols;
#pragma unroll
  for (int i = 0; i < 4; ++i)
    tile[ty + i * 8][tx] = src[(size_t)(r0 + ty + i * 8) * cols + c0 + tx];
  __syncthreads();
#pragma unroll
  for (int i = 0; i < 4; ++i)
    dst[(size_t)(c0 + ty + i * 8) * rows + r0 + tx] = (_Float16)tile[tx][ty + i * 8];
}

// ------------- per-head projection: Out = X @ W[h] + bias[h] --------------------------------
// Async-DMA double-buffered k-loop. TR=true stores Out[b,h,dk,s] (vectorized) — for V.
template <bool TR>
__global__ __launch_bounds__(256) void proj_kernel(
    const _Float16* __restrict__ X,   // [B*S, DMODEL] f16
    const _Float16* __restrict__ Wt,  // [NH, DK, DMODEL] f16 (pre-transposed)
    const float*    __restrict__ bias,// [NH, DK] f32
    _Float16*       __restrict__ Out)
{
  constexpr int LDA = 40, LDB = 40;
  __shared__ __align__(16) _Float16 sA[2][128 * LDA];
  __shared__ __align__(16) _Float16 sB[2][64 * LDB];
  const int b  = blockIdx.y >> 4;
  const int h  = blockIdx.y & 15;
  const int m0 = blockIdx.x * 128;
  const int t  = threadIdx.x;
  const int wave = t >> 5, lane = t & 31;

  const int arow = t >> 1, aseg = t & 1;      // A: 128x32, 16 halves/thread
  const int brow = t >> 2, bseg = t & 3;      // B: 64x32,  8 halves/thread
  const _Float16* aSrc = X + (size_t)(b * SS + m0 + arow) * DMODEL + aseg * 16;
  const _Float16* bSrc = Wt + ((size_t)h * DK + brow) * DMODEL + bseg * 8;
  _Float16* aDst0 = &sA[0][arow * LDA + aseg * 16];
  _Float16* bDst0 = &sB[0][brow * LDB + bseg * 8];
  _Float16* aDst1 = &sA[1][arow * LDA + aseg * 16];
  _Float16* bDst1 = &sB[1][brow * LDB + bseg * 8];

  v8f acc[4];
#pragma unroll
  for (int n = 0; n < 4; ++n)
#pragma unroll
    for (int r = 0; r < 8; ++r) acc[n][r] = 0.0f;

  async_copy_b128(aDst0, aSrc);
  async_copy_b128(aDst0 + 8, aSrc + 8);
  async_copy_b128(bDst0, bSrc);
  wait_async();
  __syncthreads();

  int cur = 0;
  for (int k0 = 0; k0 < DMODEL; k0 += 32) {
    const bool more = (k0 + 32) < DMODEL;
    if (more) {   // DMA next tile into the other buffer while WMMAs run on this one
      _Float16* ad = cur ? aDst0 : aDst1;
      _Float16* bd = cur ? bDst0 : bDst1;
      async_copy_b128(ad, aSrc + k0 + 32);
      async_copy_b128(ad + 8, aSrc + k0 + 40);
      async_copy_b128(bd, bSrc + k0 + 32);
    }
    v16h a = load_a_frag(&sA[cur][wave * 16 * LDA], LDA);
#pragma unroll
    for (int n = 0; n < 4; ++n)
      acc[n] = wmma_f16(a, load_bn_frag(&sB[cur][n * 16 * LDB], LDB), acc[n]);
    if (more) {
      wait_async();
      __syncthreads();
      cur ^= 1;
    }
  }

  const int hi = lane >> 4, nl = lane & 15;
#pragma unroll
  for (int n = 0; n < 4; ++n) {
    const int col = n * 16 + nl;
    const float bv = bias[h * DK + col];
    if (TR) {               // Out[b,h,dk,s]: 8 consecutive s per lane -> one v8h store
      v8h vo;
#pragma unroll
      for (int r = 0; r < 8; ++r) vo[r] = (_Float16)(acc[n][r] + bv);
      *(v8h*)(Out + (((size_t)b * NH + h) * DK + col) * SS + m0 + wave * 16 + 8 * hi) = vo;
    } else {
#pragma unroll
      for (int r = 0; r < 8; ++r) {
        const int srow = m0 + wave * 16 + r + 8 * hi;
        Out[(((size_t)b * NH + h) * SS + srow) * DK + col] = (_Float16)(acc[n][r] + bv);
      }
    }
  }
}

// ---------- fused scores -> batch-axis softmax -> attn @ V ----------------------------------
// Score phase: wave w = (nh=w>>2 -> t-cols nh*32..+32, sr=w&3 -> s-rows) computes BOTH batches,
// so softmax(B=2) = sigmoid(s0-s1) needs no LDS exchange: attn0=rcp(1+exp(s1-s0)), attn1=1-attn0.
// O phase: wave w = (wb=w>>2, sr) consumes attn A-fragments from LDS.
__global__ __launch_bounds__(256) void attn_kernel(
    const _Float16* __restrict__ Q,   // [B,NH,S,DK]
    const _Float16* __restrict__ Kp,  // [B,NH,S,DK]
    const _Float16* __restrict__ Vt,  // [B,NH,DK,S]  (transposed)
    _Float16*       __restrict__ O)   // [B,NH,S,DK]
{
  constexpr int LDT = 72;
  __shared__ __align__(16) _Float16 sK[2 * 64 * LDT];     // [bb][t][dk]
  __shared__ __align__(16) _Float16 sV[2 * 64 * LDT];     // [bb][dk][t]
  __shared__ __align__(16) _Float16 sAtt[8 * 16 * LDT];   // [bb][sr][m][t]

  const int h  = blockIdx.y;
  const int s0 = blockIdx.x * 64;
  const int t  = threadIdx.x;
  const int wave = t >> 5, lane = t & 31;
  const int nh = wave >> 2, sr = wave & 3;   // score-phase role
  const int wb = wave >> 2;                  // O-phase role (same bits, different meaning)
  const int hi = lane >> 4, nl = lane & 15;

  // Q fragments for BOTH batches (rows s0+sr*16), 1/sqrt(dk) folded in
  v16h qf[2][2];
#pragma unroll
  for (int bb = 0; bb < 2; ++bb) {
    const _Float16* qb = Q + (((size_t)bb * NH + h) * SS + s0 + sr * 16) * DK;
    qf[bb][0] = load_a_frag(qb, DK);
    qf[bb][1] = load_a_frag(qb + 32, DK);
#pragma unroll
    for (int e = 0; e < 16; ++e) {
      qf[bb][0][e] = qf[bb][0][e] * (_Float16)0.125f;
      qf[bb][1][e] = qf[bb][1][e] * (_Float16)0.125f;
    }
  }

  v8f o[4];
#pragma unroll
  for (int n = 0; n < 4; ++n)
#pragma unroll
    for (int r = 0; r < 8; ++r) o[n][r] = 0.0f;

  const int row = t >> 2, seg = t & 3;    // staging: 64 rows, 16 halves/thread/tensor/batch
  v8h kreg[2][2], vreg[2][2];
#pragma unroll
  for (int bb = 0; bb < 2; ++bb) {
    const size_t ks = (((size_t)bb * NH + h) * SS + row) * DK + seg * 16;
    kreg[bb][0] = *(const v8h*)(Kp + ks);
    kreg[bb][1] = *(const v8h*)(Kp + ks + 8);
    const size_t vs = (((size_t)bb * NH + h) * DK + row) * SS + seg * 16;
    vreg[bb][0] = *(const v8h*)(Vt + vs);
    vreg[bb][1] = *(const v8h*)(Vt + vs + 8);
  }

  for (int t0 = 0; t0 < SS; t0 += 64) {
    __syncthreads();                        // previous tiles fully consumed
#pragma unroll
    for (int bb = 0; bb < 2; ++bb) {
      *(v8h*)&sK[(bb * 64 + row) * LDT + seg * 16]     = kreg[bb][0];
      *(v8h*)&sK[(bb * 64 + row) * LDT + seg * 16 + 8] = kreg[bb][1];
      *(v8h*)&sV[(bb * 64 + row) * LDT + seg * 16]     = vreg[bb][0];
      *(v8h*)&sV[(bb * 64 + row) * LDT + seg * 16 + 8] = vreg[bb][1];
    }
    if (t0 + 64 < SS) {                     // prefetch next t-tile into registers
#pragma unroll
      for (int bb = 0; bb < 2; ++bb) {
        const size_t ks = (((size_t)bb * NH + h) * SS + t0 + 64 + row) * DK + seg * 16;
        kreg[bb][0] = *(const v8h*)(Kp + ks);
        kreg[bb][1] = *(const v8h*)(Kp + ks + 8);
        const size_t vs = (((size_t)bb * NH + h) * DK + row) * SS + t0 + 64 + seg * 16;
        vreg[bb][0] = *(const v8h*)(Vt + vs);
        vreg[bb][1] = *(const v8h*)(Vt + vs + 8);
      }
    }
    __syncthreads();

    // scores for both batches, t-columns [nh*32, nh*32+32)
    v8f sc[2][2];
#pragma unroll
    for (int bb = 0; bb < 2; ++bb)
#pragma unroll
      for (int j = 0; j < 2; ++j) {
        const int n = nh * 2 + j;
        v8f z;
#pragma unroll
        for (int r = 0; r < 8; ++r) z[r] = 0.0f;
        z = wmma_f16(qf[bb][0], load_bn_frag(&sK[(bb * 64 + n * 16) * LDT], LDT), z);
        sc[bb][j] = wmma_f16(qf[bb][1], load_bn_frag(&sK[(bb * 64 + n * 16) * LDT + 32], LDT), z);
      }

    // batch-pair softmax in registers; scatter attn tiles to LDS in A-consumable layout
#pragma unroll
    for (int j = 0; j < 2; ++j) {
      const int tcol = (nh * 2 + j) * 16 + nl;
#pragma unroll
      for (int r = 0; r < 8; ++r) {
        const int m = r + 8 * hi;
        const float a0 = __builtin_amdgcn_rcpf(1.0f + __expf(sc[1][j][r] - sc[0][j][r]));
        sAtt[((0 * 4 + sr) * 16 + m) * LDT + tcol] = (_Float16)a0;
        sAtt[((1 * 4 + sr) * 16 + m) * LDT + tcol] = (_Float16)(1.0f - a0);
      }
    }
    __syncthreads();

    // O += attn @ V for this wave's (wb, sr)
    v16h af0 = load_a_frag(&sAtt[((wb * 4 + sr) * 16) * LDT], LDT);
    v16h af1 = load_a_frag(&sAtt[((wb * 4 + sr) * 16) * LDT + 32], LDT);
#pragma unroll
    for (int n = 0; n < 4; ++n) {
      o[n] = wmma_f16(af0, load_bn_frag(&sV[(wb * 64 + n * 16) * LDT], LDT), o[n]);
      o[n] = wmma_f16(af1, load_bn_frag(&sV[(wb * 64 + n * 16) * LDT + 32], LDT), o[n]);
    }
  }

#pragma unroll
  for (int n = 0; n < 4; ++n)
#pragma unroll
    for (int r = 0; r < 8; ++r) {
      const int srow = s0 + sr * 16 + r + 8 * hi;
      O[(((size_t)wb * NH + h) * SS + srow) * DK + n * 16 + nl] = (_Float16)o[n][r];
    }
}

// ------------- output projection: out = concat(heads) @ Wo + bo (fp32) ----------------------
__global__ __launch_bounds__(256) void outproj_kernel(
    const _Float16* __restrict__ Oc,   // [B,NH,S,DK] (gathered as concat)
    const _Float16* __restrict__ WoT,  // [DMODEL(n), DMODEL(k)] f16 (pre-transposed)
    const float*    __restrict__ bo,   // [DMODEL]
    float*          __restrict__ out)  // [B*S, DMODEL]
{
  constexpr int LDA = 40, LDB = 40;
  __shared__ __align__(16) _Float16 sA[2][128 * LDA];
  __shared__ __align__(16) _Float16 sB[2][64 * LDB];
  const int m0 = blockIdx.x * 128;
  const int n0 = blockIdx.y * 64;
  const int t  = threadIdx.x, wave = t >> 5, lane = t & 31;

  const int arow = t >> 1, aseg = t & 1;
  const int brow = t >> 2, bseg = t & 3;
  const int g = m0 + arow;
  const int bq_ = g >> 11, s = g & (SS - 1);
  const _Float16* bSrc = WoT + (size_t)(n0 + brow) * DMODEL + bseg * 8;
  _Float16* aDst[2] = { &sA[0][arow * LDA + aseg * 16], &sA[1][arow * LDA + aseg * 16] };
  _Float16* bDst[2] = { &sB[0][brow * LDB + bseg * 8],  &sB[1][brow * LDB + bseg * 8] };

  auto aAddr = [&](int f) {   // 16-col segs never cross a head boundary
    const int hh = f >> 6, dk = f & 63;
    return Oc + (((size_t)bq_ * NH + hh) * SS + s) * DK + dk;
  };

  v8f acc[4];
#pragma unroll
  for (int n = 0; n < 4; ++n)
#pragma unroll
    for (int r = 0; r < 8; ++r) acc[n][r] = 0.0f;

  {
    const _Float16* ap = aAddr(aseg * 16);
    async_copy_b128(aDst[0], ap);
    async_copy_b128(aDst[0] + 8, ap + 8);
    async_copy_b128(bDst[0], bSrc);
    wait_async();
    __syncthreads();
  }

  int cur = 0;
  for (int k0 = 0; k0 < DMODEL; k0 += 32) {
    const bool more = (k0 + 32) < DMODEL;
    if (more) {
      const _Float16* ap = aAddr(k0 + 32 + aseg * 16);
      async_copy_b128(aDst[cur ^ 1], ap);
      async_copy_b128(aDst[cur ^ 1] + 8, ap + 8);
      async_copy_b128(bDst[cur ^ 1], bSrc + k0 + 32);
    }
    v16h a = load_a_frag(&sA[cur][wave * 16 * LDA], LDA);
#pragma unroll
    for (int n = 0; n < 4; ++n)
      acc[n] = wmma_f16(a, load_bn_frag(&sB[cur][n * 16 * LDB], LDB), acc[n]);
    if (more) {
      wait_async();
      __syncthreads();
      cur ^= 1;
    }
  }

  const int hi = lane >> 4, nl = lane & 15;
#pragma unroll
  for (int n = 0; n < 4; ++n) {
    const int col = n0 + n * 16 + nl;
    const float bv = bo[col];
#pragma unroll
    for (int r = 0; r < 8; ++r)
      out[(size_t)(m0 + wave * 16 + r + 8 * hi) * DMODEL + col] = acc[n][r] + bv;
  }
}

extern "C" void kernel_launch(void* const* d_in, const int* in_sizes, int n_in,
                              void* d_out, int out_size, void* d_ws, size_t ws_size,
                              hipStream_t stream) {
  (void)in_sizes; (void)n_in; (void)out_size; (void)ws_size;
  const float* query = (const float*)d_in[0];
  const float* key   = (const float*)d_in[1];
  const float* value = (const float*)d_in[2];
  const float* Wq = (const float*)d_in[3];
  const float* bq = (const float*)d_in[4];
  const float* Wk = (const float*)d_in[5];
  const float* bk = (const float*)d_in[6];
  const float* Wv = (const float*)d_in[7];
  const float* bv = (const float*)d_in[8];
  const float* Wo = (const float*)d_in[9];
  const float* bo = (const float*)d_in[10];
  float* out = (float*)d_out;

  const size_t XSZ = (size_t)BB * SS * DMODEL;
  const size_t WSZ = (size_t)NH * DMODEL * DK;
  const size_t QSZ = (size_t)BB * NH * SS * DK;
  _Float16* Xq   = (_Float16*)d_ws;
  _Float16* Xk   = Xq + XSZ;
  _Float16* Xv   = Xk + XSZ;
  _Float16* WqT  = Xv + XSZ;
  _Float16* WkT  = WqT + WSZ;
  _Float16* WvT  = WkT + WSZ;
  _Float16* WoT  = WvT + WSZ;
  _Float16* Qp   = WoT + (size_t)DMODEL * DMODEL;
  _Float16* Kp   = Qp + QSZ;
  _Float16* Vt   = Kp + QSZ;
  _Float16* Oc   = Vt + QSZ;

  cvt_f32_f16<<<2048, 256, 0, stream>>>(query, Xq, (int)XSZ);
  cvt_f32_f16<<<2048, 256, 0, stream>>>(key,   Xk, (int)XSZ);
  cvt_f32_f16<<<2048, 256, 0, stream>>>(value, Xv, (int)XSZ);
  cvt_transpose<<<dim3(DK / 32, DMODEL / 32, NH), 256, 0, stream>>>(Wq, WqT, DMODEL, DK);
  cvt_transpose<<<dim3(DK / 32, DMODEL / 32, NH), 256, 0, stream>>>(Wk, WkT, DMODEL, DK);
  cvt_transpose<<<dim3(DK / 32, DMODEL / 32, NH), 256, 0, stream>>>(Wv, WvT, DMODEL, DK);
  cvt_transpose<<<dim3(DMODEL / 32, DMODEL / 32, 1), 256, 0, stream>>>(Wo, WoT, DMODEL, DMODEL);

  proj_kernel<false><<<dim3(SS / 128, BB * NH), 256, 0, stream>>>(Xq, WqT, bq, Qp);
  proj_kernel<false><<<dim3(SS / 128, BB * NH), 256, 0, stream>>>(Xk, WkT, bk, Kp);
  proj_kernel<true ><<<dim3(SS / 128, BB * NH), 256, 0, stream>>>(Xv, WvT, bv, Vt);

  attn_kernel<<<dim3(SS / 64, NH), 256, 0, stream>>>(Qp, Kp, Vt, Oc);

  outproj_kernel<<<dim3((BB * SS) / 128, DMODEL / 64), 256, 0, stream>>>(Oc, WoT, bo, out);
}
